// LogitsProcessorWithLoRA_893353198250
// MI455X (gfx1250) — matrix operand order repaired and checked by reading, not compiled, not run on hardware
//
#include <hip/hip_runtime.h>
#include <hip/hip_bf16.h>

// ---------------- problem constants (from reference) ----------------
#define T_TOK 1024
#define HDIM  4096
#define NLORA 8
#define RANK  16
#define ORG_V 32000
#define EXTRA 256
#define VOCAB 32256
#define VPAD  32256
#define KLORA (NLORA * RANK)   // 128

typedef __attribute__((ext_vector_type(16))) __bf16 v16bf;
typedef __attribute__((ext_vector_type(8)))  __bf16 v8bf;
typedef __attribute__((ext_vector_type(8)))  float  v8f;
typedef unsigned int u32;
typedef __attribute__((ext_vector_type(4))) u32 v4u;
typedef __attribute__((ext_vector_type(8))) int  v8i;
typedef __attribute__((ext_vector_type(4))) int  v4i;

#if defined(__has_builtin)
#if __has_builtin(__builtin_amdgcn_tensor_load_to_lds) && \
    __has_builtin(__builtin_amdgcn_s_wait_tensorcnt)
#define USE_TDM 1
#endif
#endif

// ---------------- helpers ----------------
__device__ __forceinline__ v8bf pack8(const float4 a, const float4 b) {
    v8bf r;
    r[0] = (__bf16)a.x; r[1] = (__bf16)a.y; r[2] = (__bf16)a.z; r[3] = (__bf16)a.w;
    r[4] = (__bf16)b.x; r[5] = (__bf16)b.y; r[6] = (__bf16)b.z; r[7] = (__bf16)b.w;
    return r;
}

__device__ __forceinline__ v16bf cat16(const __bf16* lo, const __bf16* hi) {
    v8bf a = *(const v8bf*)lo;
    v8bf b = *(const v8bf*)hi;
    return __builtin_shufflevector(a, b, 0,1,2,3,4,5,6,7,8,9,10,11,12,13,14,15);
}

#ifdef USE_TDM
// Issue one TDM 2D tile load (bf16 elements): global[tile_h][tile_w] with row
// stride `row_stride` (elements) -> contiguous LDS rows of tile_w elements.
// D# layout per CDNA5 ISA section 8.3/8.4 (group0 128b, group1 256b).
// This toolchain exposes the 6-arg builtin:
//   (uint32x4 g0, int32x8 g1, int32x4 g2, int32x4 g3, int32x8 pad, i32 cpol)
__device__ __forceinline__ void tdm_load_2d_bf16(const __bf16* gptr, __bf16* lptr,
                                                 int tile_w, int tile_h, int row_stride)
{
    unsigned long long ga = (unsigned long long)(uintptr_t)gptr;
    u32 lds = (u32)(uintptr_t)lptr;   // generic LDS addr: low 32 bits = LDS byte offset

    v4u g0;
    g0[0] = 1u;                                         // count=1, user descriptor
    g0[1] = lds;                                        // lds_addr
    g0[2] = (u32)(ga & 0xFFFFFFFFull);                  // global_addr[31:0]
    g0[3] = (u32)((ga >> 32) & 0x1FFFFFFull) | (2u << 30); // global_addr[56:32] | type=2

    v8i g1;
    g1[0] = 0x10000;                                    // data_size=1 (2 bytes), mask=0
    g1[1] = (int)((u32)(tile_w & 0xFFFF) << 16);        // tensor_dim0 = tile_w
    g1[2] = (int)((u32)(tile_h & 0xFFFF) << 16);        // tensor_dim1 = tile_h
    g1[3] = (int)((u32)(tile_w & 0xFFFF) << 16);        // tile_dim0 = tile_w
    g1[4] = (int)(tile_h & 0xFFFF);                     // tile_dim1 = tile_h, tile_dim2=0
    g1[5] = row_stride;                                 // tensor_dim0_stride[31:0]
    g1[6] = 0;
    g1[7] = 0;

    v4i z4 = {0, 0, 0, 0};
    v8i z8 = {0, 0, 0, 0, 0, 0, 0, 0};
    __builtin_amdgcn_tensor_load_to_lds(g0, g1, z4, z4, z8, 0);
}
#endif

// ================= kernel 1: masked LoRA shrink =================
// XAflat[t, l*16+r] = (idx[t]==l) ? dot(x[t,:], A[l,r,:]) : 0   (bf16)
__global__ __launch_bounds__(256)
void k_shrink(const float* __restrict__ x, const float* __restrict__ A,
              const int* __restrict__ idx, __bf16* __restrict__ xa)
{
    const int t   = blockIdx.x;
    const int tid = threadIdx.x;
    const int id  = idx[t];

    __shared__ float xrow[HDIM];                       // 16 KB
    for (int i = tid; i < HDIM / 4; i += 256)
        ((float4*)xrow)[i] = ((const float4*)(x + (size_t)t * HDIM))[i];

    if (tid < KLORA) xa[(size_t)t * KLORA + tid] = (__bf16)0.0f;  // zero (mask)
    __syncthreads();

    float partial = 0.0f;
    const int r   = tid >> 4;       // 0..15
    const int sub = tid & 15;       // 0..15
    if (id >= 0) {
        const float* a = A + (((size_t)id * RANK) + r) * HDIM;
        for (int h = sub * 4; h < HDIM; h += 64) {
            float4 av = *(const float4*)(a + h);
            float4 xv = *(const float4*)(xrow + h);
            partial += av.x * xv.x + av.y * xv.y + av.z * xv.z + av.w * xv.w;
        }
    }
    for (int off = 8; off >= 1; off >>= 1)
        partial += __shfl_xor(partial, off, 16);
    if (id >= 0 && sub == 0)
        xa[(size_t)t * KLORA + id * RANK + r] = (__bf16)partial;
}

// ================= kernel 2: extra-vocab logits (per-token select) =================
__global__ __launch_bounds__(256)
void k_sel(const float* __restrict__ x, const float* __restrict__ embt,
           const int* __restrict__ idx, float* __restrict__ sel)
{
    const int t = blockIdx.x;
    const int e = threadIdx.x;
    const int id = idx[t];

    __shared__ float xrow[HDIM];
    for (int i = threadIdx.x; i < HDIM / 4; i += 256)
        ((float4*)xrow)[i] = ((const float4*)(x + (size_t)t * HDIM))[i];
    __syncthreads();

    if (id < 0) { sel[(size_t)t * EXTRA + e] = -__builtin_huge_valf(); return; }

    const float* w = embt + (((size_t)id * EXTRA) + e) * HDIM;
    float acc = 0.0f;
    for (int h = 0; h < HDIM; h += 4) {
        float4 wv = *(const float4*)(w + h);
        acc += wv.x * xrow[h] + wv.y * xrow[h+1] + wv.z * xrow[h+2] + wv.w * xrow[h+3];
    }
    sel[(size_t)t * EXTRA + e] = acc;
}

// ================= kernel 3: fused main GEMM + LoRA expand + extra-vocab merge ==========
// WG tile 128(M) x 64(N), 8 waves in 4x2, each wave owns 32x32 (four 16x16 WMMA tiles):
// 4 WMMAs per 8 ds_load_b128 fragments per K-step. Grid.x = M (fastest) so the 8
// M-blocks sharing one E column-chunk are dispatch-adjacent -> E streamed from HBM
// once, reuse through the 192MB L2. X / lora_b / sel / XA are all L2-resident.
__global__ __launch_bounds__(256)
void k_main(const float* __restrict__ X, const float* __restrict__ E,
            const float* __restrict__ Bl, const __bf16* __restrict__ XA,
            const float* __restrict__ SEL, float* __restrict__ out)
{
    __shared__ __align__(16) __bf16 As[128][32];   // A tile (bf16), 8 KB
    __shared__ __align__(16) __bf16 Bs[64][32];    // B tile (bf16), 4 KB

    const int tid  = threadIdx.x;
    const int lane = tid & 31;
    const int wave = tid >> 5;      // 0..7
    const int wm   = wave >> 1;     // 0..3 -> M offset 32*wm
    const int wn   = wave & 1;      // 0..1 -> N offset 32*wn
    const int m0   = blockIdx.x * 128;
    const int n0   = blockIdx.y * 64;

    const int half = lane >> 4;     // A frag: 0 -> K{0..15} | 1 -> K{16..31}
    const int l16  = lane & 15;

    const int srow = tid >> 2;          // 0..63
    const int scol = (tid & 3) * 8;     // 0,8,16,24

    v8f acc00 = {}, acc01 = {}, acc10 = {}, acc11 = {};   // base
    v8f dac00 = {}, dac01 = {}, dac10 = {}, dac11 = {};   // LoRA delta

    const float* xp0 = X + (size_t)(m0 + srow)      * HDIM + scol;
    const float* xp1 = X + (size_t)(m0 + 64 + srow) * HDIM + scol;
    const float* ep  = E + (size_t)(n0 + srow)      * HDIM + scol;

    // ---- base GEMM: C = X * E^T over K = HDIM ----
    for (int k0 = 0; k0 < HDIM; k0 += 32) {
        {
            float4 a = ((const float4*)(xp0 + k0))[0];
            float4 b = ((const float4*)(xp0 + k0))[1];
            *(v8bf*)&As[srow][scol] = pack8(a, b);
            a = ((const float4*)(xp1 + k0))[0];
            b = ((const float4*)(xp1 + k0))[1];
            *(v8bf*)&As[64 + srow][scol] = pack8(a, b);
            a = ((const float4*)(ep + k0))[0];
            b = ((const float4*)(ep + k0))[1];
            *(v8bf*)&Bs[srow][scol] = pack8(a, b);
            if (k0 + 32 < HDIM) {
                __builtin_prefetch(ep  + k0 + 32, 0, 0);   // global_prefetch_b8
                __builtin_prefetch(xp0 + k0 + 32, 0, 0);
                __builtin_prefetch(xp1 + k0 + 32, 0, 0);
            }
        }
        __syncthreads();

        v16bf a0 = *(const v16bf*)&As[wm * 32      + l16][half * 16];
        v16bf a1 = *(const v16bf*)&As[wm * 32 + 16 + l16][half * 16];
        v16bf b0 = *(const v16bf*)&Bs[wn * 32      + l16][half * 16];
        v16bf b1 = *(const v16bf*)&Bs[wn * 32 + 16 + l16][half * 16];

        acc00 = __builtin_amdgcn_wmma_f32_16x16x32_bf16(false, a0, false, b0, (short)0, acc00, false, false);
        acc01 = __builtin_amdgcn_wmma_f32_16x16x32_bf16(false, a0, false, b1, (short)0, acc01, false, false);
        acc10 = __builtin_amdgcn_wmma_f32_16x16x32_bf16(false, a1, false, b0, (short)0, acc10, false, false);
        acc11 = __builtin_amdgcn_wmma_f32_16x16x32_bf16(false, a1, false, b1, (short)0, acc11, false, false);
        __syncthreads();
    }

    // ---- LoRA expand as GEMM over flattened K = L*R = 128 ----
    // delta = XAflat[T,128] @ Bflat[VPAD,128]^T, Bflat[v, l*16+r] = lora_b[l, v, r]
    for (int kb = 0; kb < KLORA; kb += 32) {
#ifdef USE_TDM
        // A tile is already bf16 in memory: raw 2D copy -> Tensor Data Mover.
        if (wave == 0)
            tdm_load_2d_bf16(XA + (size_t)m0 * KLORA + kb, &As[0][0],
                             /*tile_w=*/32, /*tile_h=*/128, /*row_stride=*/KLORA);
#else
        *(v8bf*)&As[srow][scol] =
            *(const v8bf*)(XA + (size_t)(m0 + srow) * KLORA + kb + scol);
        *(v8bf*)&As[64 + srow][scol] =
            *(const v8bf*)(XA + (size_t)(m0 + 64 + srow) * KLORA + kb + scol);
#endif
        {
            const int c = kb + scol;           // flattened K of first element
            const int l = c >> 4;              // lora id
            const int r = c & 15;              // rank offset (0 or 8)
            const float* q = Bl + (((size_t)l * VPAD) + (n0 + srow)) * RANK + r;
            float4 a = ((const float4*)q)[0];
            float4 b = ((const float4*)q)[1];
            *(v8bf*)&Bs[srow][scol] = pack8(a, b);
        }
#ifdef USE_TDM
        if (wave == 0) __builtin_amdgcn_s_wait_tensorcnt(0);
#endif
        __syncthreads();

        v16bf a0 = *(const v16bf*)&As[wm * 32      + l16][half * 16];
        v16bf a1 = *(const v16bf*)&As[wm * 32 + 16 + l16][half * 16];
        v16bf b0 = *(const v16bf*)&Bs[wn * 32      + l16][half * 16];
        v16bf b1 = *(const v16bf*)&Bs[wn * 32 + 16 + l16][half * 16];

        dac00 = __builtin_amdgcn_wmma_f32_16x16x32_bf16(false, a0, false, b0, (short)0, dac00, false, false);
        dac01 = __builtin_amdgcn_wmma_f32_16x16x32_bf16(false, a0, false, b1, (short)0, dac01, false, false);
        dac10 = __builtin_amdgcn_wmma_f32_16x16x32_bf16(false, a1, false, b0, (short)0, dac10, false, false);
        dac11 = __builtin_amdgcn_wmma_f32_16x16x32_bf16(false, a1, false, b1, (short)0, dac11, false, false);
        __syncthreads();
    }

    // ---- epilogue: C/D layout VGPR j -> (M = j + 8*half, N = l16) ----
#pragma unroll
    for (int bi = 0; bi < 2; ++bi) {
        const int col = n0 + wn * 32 + bi * 16 + l16;
        const bool extra = (col >= ORG_V);
        const int eoff = col - ORG_V;
        const v8f& aT = bi ? acc01 : acc00;   // top 16 rows of wave tile
        const v8f& aB = bi ? acc11 : acc10;   // bottom 16 rows
        const v8f& dT = bi ? dac01 : dac00;
        const v8f& dB = bi ? dac11 : dac10;
#pragma unroll
        for (int j = 0; j < 8; ++j) {
            const int mA = m0 + wm * 32 + j + half * 8;
            const int mB = mA + 16;
            float vA = extra ? (SEL[(size_t)mA * EXTRA + eoff] + dT[j]) : (aT[j] + dT[j]);
            float vB = extra ? (SEL[(size_t)mB * EXTRA + eoff] + dB[j]) : (aB[j] + dB[j]);
            out[(size_t)mA * VOCAB + col] = vA;
            out[(size_t)mB * VOCAB + col] = vB;
        }
    }
}

// ================= launcher =================
extern "C" void kernel_launch(void* const* d_in, const int* in_sizes, int n_in,
                              void* d_out, int out_size, void* d_ws, size_t ws_size,
                              hipStream_t stream) {
    const float* x    = (const float*)d_in[0];   // [T, H]
    const float* emb  = (const float*)d_in[1];   // [VOCAB, H]
    const float* embt = (const float*)d_in[2];   // [L, EXTRA, H]
    const float* la   = (const float*)d_in[3];   // [L, R, H]
    const float* lb   = (const float*)d_in[4];   // [L, VPAD, R]
    const int*   idx  = (const int*)d_in[5];     // [T]
    float* out = (float*)d_out;                  // [T, VOCAB]

    float*  sel = (float*)d_ws;                                        // T*EXTRA f32 (1 MB)
    __bf16* xa  = (__bf16*)((char*)d_ws + (size_t)T_TOK * EXTRA * 4);  // T*128 bf16 (256 KB)

    k_shrink<<<T_TOK, 256, 0, stream>>>(x, la, idx, xa);
    k_sel   <<<T_TOK, 256, 0, stream>>>(x, embt, idx, sel);

    dim3 grid(T_TOK / 128, VPAD / 64);   // M fastest -> E column-chunk reuse in L2
    k_main<<<grid, 256, 0, stream>>>(x, emb, lb, xa, sel, out);
}
